// AttentionHead_84765474554569
// MI455X (gfx1250) — matrix operand order repaired and checked by reading, not compiled, not run on hardware
//
#include <hip/hip_runtime.h>
#include <hip/hip_bf16.h>

// ---------------------------------------------------------------------------
// Single-head causal attention, B=8 T=4096 D=1024 H=64, fp32 in/out.
//   (1) W fp32->bf16
//   (2) QKV projection via v_wmma bf16 (V stored transposed)
//   (3) flash attention: K/V tiles staged in LDS via async loads (ASYNCcnt,
//       double-buffered), shared by 4 waves/workgroup; matmuls on v_wmma bf16.
// ---------------------------------------------------------------------------

#define B_ 8
#define T_ 4096
#define D_ 1024
#define H_ 64

typedef __attribute__((ext_vector_type(16))) __bf16 bf16x16;
typedef __attribute__((ext_vector_type(8)))  __bf16 bf16x8;
typedef __attribute__((ext_vector_type(8)))  float  f32x8;
typedef __attribute__((ext_vector_type(4)))  float  f32x4;

static __device__ inline f32x8 zero8() {
    f32x8 z;
#pragma unroll
    for (int i = 0; i < 8; ++i) z[i] = 0.0f;
    return z;
}

// A-matrix (16x32 bf16) per-lane fragment: lane = M in each
// 16-lane half h; elements 0..15 -> K = h*16+e (matches B operands).
static __device__ inline bf16x16 load_a_bf16(const __bf16* row, int h) {
    bf16x8 lo = *(const bf16x8*)(row + h * 16);
    bf16x8 hi = *(const bf16x8*)(row + h * 16 + 8);
    bf16x16 r;
#pragma unroll
    for (int e = 0; e < 8; ++e) { r[e] = lo[e]; r[8 + e] = hi[e]; }
    return r;
}

// Same A fragment sourced from fp32 (x matrix), converted on the fly.
static __device__ inline bf16x16 load_a_f32(const float* __restrict__ row, int h) {
    const f32x4* p0 = (const f32x4*)(row + h * 16);
    const f32x4* p1 = (const f32x4*)(row + h * 16 + 8);
    f32x4 a0 = p0[0], a1 = p0[1], b0 = p1[0], b1 = p1[1];
    bf16x16 r;
#pragma unroll
    for (int e = 0; e < 4; ++e) {
        r[e]      = (__bf16)a0[e];
        r[4 + e]  = (__bf16)a1[e];
        r[8 + e]  = (__bf16)b0[e];
        r[12 + e] = (__bf16)b1[e];
    }
    return r;
}

// Reductions across a 16-lane half (C-matrix rows: lanes 0-15 hold M=0..7,
// lanes 16-31 hold M=8..15; xor masks <=8 never cross halves).
static __device__ inline float half_max(float v) {
#pragma unroll
    for (int mk = 1; mk <= 8; mk <<= 1) v = fmaxf(v, __shfl_xor(v, mk, 32));
    return v;
}
static __device__ inline float half_sum(float v) {
#pragma unroll
    for (int mk = 1; mk <= 8; mk <<= 1) v += __shfl_xor(v, mk, 32);
    return v;
}

// Async global->LDS copy of one 16-byte chunk (CDNA5 ASYNCcnt path).
static __device__ inline void async_copy_b128(unsigned lds_byte_addr,
                                              const __bf16* sbase,
                                              int voff_bytes) {
    asm volatile("global_load_async_to_lds_b128 %0, %1, %2 offset:0"
                 :
                 : "v"(lds_byte_addr), "v"(voff_bytes), "s"(sbase)
                 : "memory");
}
#define ASYNC_WAIT_4() asm volatile("s_wait_asynccnt 0x4" ::: "memory")
#define ASYNC_WAIT_0() asm volatile("s_wait_asynccnt 0x0" ::: "memory")

// ---------------------------------------------------------------------------
// Kernel 1: W_q|W_k|W_v fp32 -> one bf16 array [3][64*1024]
// ---------------------------------------------------------------------------
__global__ void __launch_bounds__(256)
convert_w_kernel(const float* __restrict__ wq, const float* __restrict__ wk,
                 const float* __restrict__ wv, __bf16* __restrict__ dst) {
    int i = blockIdx.x * 256 + threadIdx.x;        // [0, 3*65536)
    int which = i >> 16;
    int j = i & 65535;
    const float* src = (which == 0) ? wq : (which == 1) ? wk : wv;
    dst[i] = (__bf16)src[j];
}

// ---------------------------------------------------------------------------
// Kernel 2: QKV projection. One wave computes a 16-row tile of Q,K,V.
//   Q,K row-major bf16 [B*T, 64]; V transposed bf16 [B, 64, T].
// ---------------------------------------------------------------------------
__global__ void __launch_bounds__(128)
proj_qkv_kernel(const float* __restrict__ x, const __bf16* __restrict__ wbf,
                __bf16* __restrict__ qbf, __bf16* __restrict__ kbf,
                __bf16* __restrict__ vtbf) {
    const int lane = threadIdx.x & 31;
    const int wid  = threadIdx.x >> 5;
    const int rt   = blockIdx.x * 4 + wid;     // row tile id, [0, 2048)
    const int b    = rt >> 8;
    const int t0   = (rt & 255) << 4;
    const int m    = lane & 15;
    const int h    = lane >> 4;
    const int n16  = lane & 15;

    f32x8 acc[3][4];
#pragma unroll
    for (int w = 0; w < 3; ++w)
#pragma unroll
        for (int nt = 0; nt < 4; ++nt) acc[w][nt] = zero8();

    const float* xrow = x + ((size_t)b * T_ + t0 + m) * D_;

    for (int ks = 0; ks < D_ / 32; ++ks) {
        bf16x16 a = load_a_f32(xrow + ks * 32, h);
#pragma unroll
        for (int w = 0; w < 3; ++w) {
            bf16x16 bv[4];
#pragma unroll
            for (int nt = 0; nt < 4; ++nt) {
                // B-matrix 32x16: lane = N, elements -> K = h*16 + e
                const __bf16* wp = wbf + (size_t)w * (H_ * D_) +
                                   (size_t)(nt * 16 + n16) * D_ + ks * 32 + h * 16;
                bv[nt] = *(const bf16x16*)wp;
            }
#pragma unroll
            for (int nt = 0; nt < 4; ++nt)
                acc[w][nt] = __builtin_amdgcn_wmma_f32_16x16x32_bf16(
                    false, a, false, bv[nt], (short)0, acc[w][nt], false, false);
        }
    }

    // C layout: VGPR r of lane -> (M = r + 8*h, N = lane%16)
#pragma unroll
    for (int nt = 0; nt < 4; ++nt) {
#pragma unroll
        for (int r = 0; r < 8; ++r) {
            int row = t0 + r + 8 * h;
            int col = nt * 16 + n16;
            size_t rm = ((size_t)b * T_ + row) * H_ + col;
            qbf[rm] = (__bf16)acc[0][nt][r];
            kbf[rm] = (__bf16)acc[1][nt][r];
            vtbf[((size_t)b * H_ + col) * T_ + row] = (__bf16)acc[2][nt][r];
        }
    }
}

// ---------------------------------------------------------------------------
// Kernel 3: flash attention. 4 waves/workgroup own q-tiles q0..q0+63 of one
// batch; each 32-key K tile (32x64 bf16) and V^T tile (64x32 bf16) is staged
// once in LDS via async loads (double-buffered) and consumed by all 4 waves.
// ---------------------------------------------------------------------------
__global__ void __launch_bounds__(128)
attn_kernel(const __bf16* __restrict__ qbf, const __bf16* __restrict__ kbf,
            const __bf16* __restrict__ vtbf, float* __restrict__ out) {
    __shared__ __bf16 ktile[2][32 * 64];       // [buf][key][hdim]   4KB each
    __shared__ __bf16 vtile[2][64 * 32];       // [buf][hdim][key]   4KB each
    __shared__ __bf16 pbuf[4][16 * 32];        // per-wave P staging

    const int tid  = threadIdx.x;
    const int lane = tid & 31;
    const int wid  = tid >> 5;
    const int bt   = blockIdx.x;               // [0, 512): 64-query block
    const int b    = bt >> 6;                  // 64 blocks per batch
    const int q0   = ((bt & 63) << 6) + wid * 16;
    const int n16  = lane & 15;
    const int h    = lane >> 4;
    const int m    = lane & 15;

    const __bf16* kbase = kbf + (size_t)b * T_ * H_;
    const __bf16* vbase = vtbf + (size_t)b * H_ * T_;

    // Q tile (16x64) in A-layout, two K=32 fragments, kept in VGPRs.
    const __bf16* qrow = qbf + ((size_t)b * T_ + q0 + m) * H_;
    bf16x16 qa0 = load_a_bf16(qrow, h);
    bf16x16 qa1 = load_a_bf16(qrow + 32, h);

    f32x8 o[4];
#pragma unroll
    for (int nt = 0; nt < 4; ++nt) o[nt] = zero8();
    float mrow[8], lrow[8];
#pragma unroll
    for (int r = 0; r < 8; ++r) { mrow[r] = -1e30f; lrow[r] = 0.0f; }

    const int nblk = (q0 + 16 + 31) >> 5;              // my causal clip
    const int q0w3 = ((bt & 63) << 6) + 48;
    const int nmax = (q0w3 + 16 + 31) >> 5;            // workgroup max
    __bf16* pw = &pbuf[wid][0];

    // Per-iteration LDS fill: 128 threads x (2 K-chunks + 2 V-chunks) of 16B.
    // K chunk c (0..255): row r=c>>3, part p=c&7 -> mem (k0+r)*64+p*8.
    // V chunk c (0..255): row n=c>>2, part p=c&3 -> mem n*T + k0 + p*8.
    const int kc0 = tid, kc1 = tid + 128;
    auto issue_fill = [&](int k0, int nb) {
        unsigned kl = (unsigned)(uintptr_t)(&ktile[nb][0]);
        unsigned vl = (unsigned)(uintptr_t)(&vtile[nb][0]);
        async_copy_b128(kl + (unsigned)kc0 * 16, kbase,
                        ((k0 + (kc0 >> 3)) * 64 + (kc0 & 7) * 8) * 2);
        async_copy_b128(kl + (unsigned)kc1 * 16, kbase,
                        ((k0 + (kc1 >> 3)) * 64 + (kc1 & 7) * 8) * 2);
        async_copy_b128(vl + (unsigned)kc0 * 16, vbase,
                        ((kc0 >> 2) * T_ + k0 + (kc0 & 3) * 8) * 2);
        async_copy_b128(vl + (unsigned)kc1 * 16, vbase,
                        ((kc1 >> 2) * T_ + k0 + (kc1 & 3) * 8) * 2);
    };

    issue_fill(0, 0);                                   // prologue

    for (int kb = 0; kb < nmax; ++kb) {
        const int k0 = kb << 5;
        const int cur = kb & 1;
        if (kb + 1 < nmax) {
            issue_fill((kb + 1) << 5, cur ^ 1);
            ASYNC_WAIT_4();        // in-order: cur's 4 copies complete
        } else {
            ASYNC_WAIT_0();
        }
        __syncthreads();           // cur tile visible to all waves

        if (kb < nblk) {           // wave-uniform causal clip (EXEC stays full)
            // S = Q K^T : two 16x16 tiles from LDS K
            f32x8 s[2];
#pragma unroll
            for (int nt2 = 0; nt2 < 2; ++nt2) {
                const __bf16* krow = &ktile[cur][(nt2 * 16 + n16) * 64 + h * 16];
                bf16x16 b0 = *(const bf16x16*)(krow);
                bf16x16 b1 = *(const bf16x16*)(krow + 32);
                f32x8 a = zero8();
                a = __builtin_amdgcn_wmma_f32_16x16x32_bf16(
                        false, qa0, false, b0, (short)0, a, false, false);
                a = __builtin_amdgcn_wmma_f32_16x16x32_bf16(
                        false, qa1, false, b1, (short)0, a, false, false);
                s[nt2] = a;
            }

            const bool boundary = (k0 + 31 > q0);
#pragma unroll
            for (int r = 0; r < 8; ++r) {
                float s0 = s[0][r] * 0.125f;           // 1/sqrt(64)
                float s1 = s[1][r] * 0.125f;
                if (boundary) {
                    int mg = q0 + r + 8 * h;
                    s0 = (k0 + n16 > mg)      ? -1e30f : s0;
                    s1 = (k0 + 16 + n16 > mg) ? -1e30f : s1;
                }
                float rmax = half_max(fmaxf(s0, s1));
                float mnew = fmaxf(mrow[r], rmax);
                float corr = __expf(mrow[r] - mnew);
                float p0 = __expf(s0 - mnew);
                float p1 = __expf(s1 - mnew);
                lrow[r] = lrow[r] * corr + half_sum(p0 + p1);
                mrow[r] = mnew;
#pragma unroll
                for (int nt = 0; nt < 4; ++nt) o[nt][r] *= corr;
                int mr = r + 8 * h;
                pw[mr * 32 + n16]      = (__bf16)p0;   // C-layout -> [16][32]
                pw[mr * 32 + 16 + n16] = (__bf16)p1;
            }
            __builtin_amdgcn_wave_barrier();
            bf16x16 pa = load_a_bf16(pw + m * 32, h);  // A-layout reread
            __builtin_amdgcn_wave_barrier();

            // O += P (16x32) @ V (32x64) from LDS V^T
#pragma unroll
            for (int nt = 0; nt < 4; ++nt) {
                const __bf16* vp = &vtile[cur][(nt * 16 + n16) * 32 + h * 16];
                bf16x16 bv = *(const bf16x16*)vp;
                o[nt] = __builtin_amdgcn_wmma_f32_16x16x32_bf16(
                    false, pa, false, bv, (short)0, o[nt], false, false);
            }
        }
        __syncthreads();           // protect cur buffer before refill at kb+2
    }

    // Epilogue: normalize rows, store fp32 (coalesced across each half).
#pragma unroll
    for (int r = 0; r < 8; ++r) {
        float inv = 1.0f / lrow[r];
        int row = q0 + r + 8 * h;
#pragma unroll
        for (int nt = 0; nt < 4; ++nt)
            out[((size_t)b * T_ + row) * H_ + nt * 16 + n16] = o[nt][r] * inv;
    }
}

// ---------------------------------------------------------------------------
extern "C" void kernel_launch(void* const* d_in, const int* in_sizes, int n_in,
                              void* d_out, int out_size, void* d_ws, size_t ws_size,
                              hipStream_t stream) {
    const float* x  = (const float*)d_in[0];
    const float* Wq = (const float*)d_in[1];
    const float* Wk = (const float*)d_in[2];
    const float* Wv = (const float*)d_in[3];
    float* out = (float*)d_out;

    // Workspace: [0,384K) W bf16; then Q, K, V^T bf16 (4MB each).
    char* ws = (char*)d_ws;
    __bf16* wbf  = (__bf16*)(ws);
    __bf16* qbf  = (__bf16*)(ws + 393216);
    __bf16* kbf  = (__bf16*)(ws + 393216 + (size_t)B_ * T_ * H_ * 2);
    __bf16* vtbf = (__bf16*)(ws + 393216 + (size_t)2 * B_ * T_ * H_ * 2);

    convert_w_kernel<<<768, 256, 0, stream>>>(Wq, Wk, Wv, wbf);
    proj_qkv_kernel<<<(B_ * T_ / 16) / 4, 128, 0, stream>>>(x, wbf, qbf, kbf, vtbf);
    attn_kernel<<<B_ * (T_ / 64), 128, 0, stream>>>(qbf, kbf, vtbf, out);
}
